// LatentMoE_28252294873414
// MI455X (gfx1250) — matrix operand order repaired and checked by reading, not compiled, not run on hardware
//
#include <hip/hip_runtime.h>
#include <hip/hip_bf16.h>
#include <stdint.h>

// ---------------- model constants ----------------
#define DIMD    1024
#define LATENTD 512
#define INTERD  512
#define SHAREDD 4096
#define NE      8
#define TOPKK   2
#define NTOK    2048          // B*S
#define NSLOT   4096          // NTOK*TOPK

typedef __attribute__((ext_vector_type(16))) __bf16 v16bf;
typedef __attribute__((ext_vector_type(8)))  float  v8f;

union FragU { uint4 u[2]; v16bf v; };

__device__ __forceinline__ unsigned short f32_to_bf16_bits(float f) {
    unsigned int u = __float_as_uint(f);
    u += 0x7FFFu + ((u >> 16) & 1u);      // round-to-nearest-even
    return (unsigned short)(u >> 16);
}

#define WMMA_BF16(a, b, c) \
    __builtin_amdgcn_wmma_f32_16x16x32_bf16(false, (a), false, (b), (short)0, (c), false, false)

// ---------------- fp32 -> bf16 bulk convert ----------------
__global__ void k_cvt_bf16(const float* __restrict__ s, unsigned short* __restrict__ d, int n) {
    int i = blockIdx.x * blockDim.x + threadIdx.x;
    if (i < n) d[i] = f32_to_bf16_bits(s[i]);
}

// ---------------- router: sigmoid scores + group-limited top-k ----------------
// one wave (32 lanes) per token
__global__ __launch_bounds__(32) void k_router(const float* __restrict__ x,
                                               const float* __restrict__ gate,
                                               const float* __restrict__ bias,
                                               int* __restrict__ sel,
                                               float* __restrict__ wts) {
    int n = blockIdx.x;
    int lane = threadIdx.x;
    const float* xr = x + n * DIMD;
    float acc[NE];
#pragma unroll
    for (int e = 0; e < NE; ++e) acc[e] = 0.f;
    for (int k = lane; k < DIMD; k += 32) {
        float xv = xr[k];
#pragma unroll
        for (int e = 0; e < NE; ++e) acc[e] += xv * gate[e * DIMD + k];
    }
#pragma unroll
    for (int e = 0; e < NE; ++e)
        for (int off = 16; off > 0; off >>= 1) acc[e] += __shfl_xor(acc[e], off, 32);

    if (lane == 0) {
        float s[NE], sfc[NE];
#pragma unroll
        for (int e = 0; e < NE; ++e) { s[e] = 1.f / (1.f + __expf(-acc[e])); sfc[e] = s[e] + bias[e]; }
        // group score = sum of top-2 within each group of 4 experts
        float gsc[2];
#pragma unroll
        for (int g = 0; g < 2; ++g) {
            float m1 = -1e30f, m2 = -1e30f;
#pragma unroll
            for (int j = 0; j < 4; ++j) {
                float v = sfc[g * 4 + j];
                if (v > m1) { m2 = m1; m1 = v; } else if (v > m2) m2 = v;
            }
            gsc[g] = m1 + m2;
        }
        int g = (gsc[1] > gsc[0]) ? 1 : 0;   // top-1 group (ties -> lower index)
        int e1 = -1, e2 = -1; float b1 = -1e30f, b2 = -1e30f;
#pragma unroll
        for (int j = 0; j < 4; ++j) {
            int e = g * 4 + j; float v = sfc[e];
            if (v > b1) { b2 = b1; e2 = e1; b1 = v; e1 = e; }
            else if (v > b2) { b2 = v; e2 = e; }
        }
        float s1 = s[e1], s2 = s[e2];
        float inv = 1.f / (s1 + s2 + 1e-20f);   // NORM_TOPK
        sel[n * 2] = e1;  sel[n * 2 + 1] = e2;
        wts[n * 2] = s1 * inv; wts[n * 2 + 1] = s2 * inv;
    }
}

// ---------------- deterministic stable counting sort by expert id ----------------
__global__ __launch_bounds__(32) void k_sort(const int* __restrict__ sel,
                                             const float* __restrict__ wts,
                                             int* __restrict__ offs,
                                             int* __restrict__ slot_tok,
                                             float* __restrict__ slot_score,
                                             int* __restrict__ tok_slot) {
    int lane = threadIdx.x;
    int cnt = 0;
    if (lane < NE) for (int i = 0; i < NSLOT; ++i) if (sel[i] == lane) cnt++;
    int off = 0;
    for (int e = 0; e < NE; ++e) {
        int c = __shfl(cnt, e, 32);
        if (e < lane) off += c;
    }
    if (lane < NE) offs[lane] = off;
    if (lane == 0) offs[NE] = NSLOT;
    if (lane < NE) {
        int p = off;
        for (int i = 0; i < NSLOT; ++i) {
            if (sel[i] == lane) {
                slot_tok[p] = i >> 1;        // token = flat_index / TOPK
                slot_score[p] = wts[i];
                tok_slot[i] = p;             // inverse map for deterministic combine
                p++;
            }
        }
    }
}

// ---------------- dispatch: xs[slot] = lat_bf16[slot_tok[slot]] ----------------
__global__ void k_dispatch(const unsigned int* __restrict__ latb,
                           const int* __restrict__ slot_tok,
                           unsigned int* __restrict__ xs) {
    int idx = blockIdx.x * blockDim.x + threadIdx.x;   // NSLOT * 256
    int slot = idx >> 8, c = idx & 255;
    int tok = slot_tok[slot];
    xs[slot * 256 + c] = latb[tok * 256 + c];
}

// ---------------- combine: comb[n] = out_slots[slotA] + out_slots[slotB] (bf16) ----------------
__global__ void k_combine(const float* __restrict__ outslots,
                          const int* __restrict__ tok_slot,
                          unsigned short* __restrict__ combb) {
    int idx = blockIdx.x * blockDim.x + threadIdx.x;   // NTOK * 512
    int n = idx >> 9, c = idx & 511;
    int s0 = tok_slot[n * 2], s1 = tok_slot[n * 2 + 1];
    combb[idx] = f32_to_bf16_bits(outslots[s0 * 512 + c] + outslots[s1 * 512 + c]);
}

// ---------------- WMMA fragment loads (wave32 layouts per CDNA5 ISA 7.12.2) ----------------
// A 16x32 bf16: lane holds row = m0 + (lane&15); elems 0-7 -> K = k0 + (lane>>4)*8 + j,
//               elems 8-15 -> K = k0 + 16 + (lane>>4)*8 + j   (two 16B runs)
__device__ __forceinline__ v16bf ldA(const unsigned short* A, int lda, int row, int k0, int lane) {
    const unsigned short* p = A + row * lda + k0 + ((lane >> 4) << 3);
    FragU f;
    f.u[0] = *(const uint4*)p;
    f.u[1] = *(const uint4*)(p + 16);
    return f.v;
}
// B 32x16 bf16: lane holds col = n0 + (lane&15); elems 0-7 -> K = k0 + (lane>>4)*8 + j,
//               elems 8-15 -> K = k0 + 16 + (lane>>4)*8 + j   (two 16B runs)
// col of B == row of W (row-major [Nout x K]) since we compute x @ W^T  -> contiguous.
__device__ __forceinline__ v16bf ldB(const unsigned short* W, int ldw, int col, int k0, int lane) {
    const unsigned short* p = W + col * ldw + k0 + ((lane >> 4) << 3);
    FragU f;
    f.u[0] = *(const uint4*)p;
    f.u[1] = *(const uint4*)(p + 16);
    return f.v;
}

// ---------------- generic GEMM: OUT[M x Nout] = A_bf16[M x K] @ W_bf16[Nout x K]^T ----------------
// Register-blocked: block = 4 waves = 64(M) x 128(N) macro tile, each wave 32(M) x 64(N):
// 2 A frags x 4 B frags -> 8 WMMAs / K-step, 8 v8f accumulators.
// EPI: 0 = bf16 store, 1 = relu^2 then bf16 store, 2 = f32 store, 3 = f32 accumulate (+= v*scale)
template <int EPI>
__global__ __launch_bounds__(128) void k_gemm(const unsigned short* __restrict__ A, int lda,
                                              const unsigned short* __restrict__ W, int ldw,
                                              int Mrows, int K, int ldo,
                                              unsigned short* __restrict__ outB,
                                              float* __restrict__ outF, float scale) {
    int lane = threadIdx.x & 31, wave = threadIdx.x >> 5;
    int l15 = lane & 15;
    int n0 = blockIdx.x * 128 + (wave & 1) * 64;
    int m0 = blockIdx.y * 64 + (wave >> 1) * 32;
    if (m0 >= Mrows) return;
    int arow0 = m0 + l15;      if (arow0 >= Mrows) arow0 = Mrows - 1;
    int arow1 = m0 + 16 + l15; if (arow1 >= Mrows) arow1 = Mrows - 1;
    int bcol = n0 + l15;
    v8f c[2][4] = {};
    for (int k0 = 0; k0 < K; k0 += 32) {
        v16bf a0 = ldA(A, lda, arow0, k0, lane);
        v16bf a1 = ldA(A, lda, arow1, k0, lane);
        v16bf b0 = ldB(W, ldw, bcol,      k0, lane);
        v16bf b1 = ldB(W, ldw, bcol + 16, k0, lane);
        v16bf b2 = ldB(W, ldw, bcol + 32, k0, lane);
        v16bf b3 = ldB(W, ldw, bcol + 48, k0, lane);
        c[0][0] = WMMA_BF16(a0, b0, c[0][0]);
        c[0][1] = WMMA_BF16(a0, b1, c[0][1]);
        c[0][2] = WMMA_BF16(a0, b2, c[0][2]);
        c[0][3] = WMMA_BF16(a0, b3, c[0][3]);
        c[1][0] = WMMA_BF16(a1, b0, c[1][0]);
        c[1][1] = WMMA_BF16(a1, b1, c[1][1]);
        c[1][2] = WMMA_BF16(a1, b2, c[1][2]);
        c[1][3] = WMMA_BF16(a1, b3, c[1][3]);
    }
    // C layout: lanes 0-15 -> M = sub_m + r, lanes 16-31 -> M = sub_m + 8 + r; N = sub_n + (lane&15)
    int mo = (lane >> 4) << 3;
#pragma unroll
    for (int i = 0; i < 2; ++i) {
#pragma unroll
        for (int j = 0; j < 4; ++j) {
            int col = n0 + j * 16 + l15;
#pragma unroll
            for (int r = 0; r < 8; ++r) {
                int m = m0 + i * 16 + mo + r;
                if (m >= Mrows) continue;
                float v = c[i][j][r];
                if (EPI == 0)      outB[m * ldo + col] = f32_to_bf16_bits(v);
                else if (EPI == 1) { float t = v > 0.f ? v : 0.f; outB[m * ldo + col] = f32_to_bf16_bits(t * t); }
                else if (EPI == 2) outF[m * ldo + col] = v;
                else               outF[m * ldo + col] += v * scale;
            }
        }
    }
}

// ---------------- grouped expert GEMM (512x512 per expert, ragged rows) ----------------
// Same 32x64 wave tiling. EPI: 1 = relu^2 -> bf16 (w1) ; 4 = f32 * per-row routing score (w2)
template <int EPI>
__global__ __launch_bounds__(128) void k_gemm_grouped(const unsigned short* __restrict__ A,
                                                      const unsigned short* __restrict__ Wall,
                                                      const int* __restrict__ offs,
                                                      unsigned short* __restrict__ outB,
                                                      float* __restrict__ outF,
                                                      const float* __restrict__ rowscale) {
    int e = blockIdx.z;
    int base = offs[e], rows = offs[e + 1] - base;
    int lane = threadIdx.x & 31, wave = threadIdx.x >> 5;
    int l15 = lane & 15;
    int n0 = blockIdx.x * 128 + (wave & 1) * 64;
    int m0 = blockIdx.y * 64 + (wave >> 1) * 32;
    if (m0 >= rows) return;
    const unsigned short* W = Wall + e * (512 * 512);
    int aml0 = m0 + l15;      if (aml0 >= rows) aml0 = rows - 1;
    int aml1 = m0 + 16 + l15; if (aml1 >= rows) aml1 = rows - 1;
    int arow0 = base + aml0, arow1 = base + aml1;
    int bcol = n0 + l15;
    v8f c[2][4] = {};
    for (int k0 = 0; k0 < 512; k0 += 32) {
        v16bf a0 = ldA(A, 512, arow0, k0, lane);
        v16bf a1 = ldA(A, 512, arow1, k0, lane);
        v16bf b0 = ldB(W, 512, bcol,      k0, lane);
        v16bf b1 = ldB(W, 512, bcol + 16, k0, lane);
        v16bf b2 = ldB(W, 512, bcol + 32, k0, lane);
        v16bf b3 = ldB(W, 512, bcol + 48, k0, lane);
        c[0][0] = WMMA_BF16(a0, b0, c[0][0]);
        c[0][1] = WMMA_BF16(a0, b1, c[0][1]);
        c[0][2] = WMMA_BF16(a0, b2, c[0][2]);
        c[0][3] = WMMA_BF16(a0, b3, c[0][3]);
        c[1][0] = WMMA_BF16(a1, b0, c[1][0]);
        c[1][1] = WMMA_BF16(a1, b1, c[1][1]);
        c[1][2] = WMMA_BF16(a1, b2, c[1][2]);
        c[1][3] = WMMA_BF16(a1, b3, c[1][3]);
    }
    int mo = (lane >> 4) << 3;
#pragma unroll
    for (int i = 0; i < 2; ++i) {
#pragma unroll
        for (int j = 0; j < 4; ++j) {
            int col = n0 + j * 16 + l15;
#pragma unroll
            for (int r = 0; r < 8; ++r) {
                int ml = m0 + i * 16 + mo + r;
                if (ml >= rows) continue;
                int g = (base + ml) * 512 + col;
                float v = c[i][j][r];
                if (EPI == 1) { float t = v > 0.f ? v : 0.f; outB[g] = f32_to_bf16_bits(t * t); }
                else          outF[g] = v * rowscale[base + ml];
            }
        }
    }
}

// ---------------- host side ----------------
extern "C" void kernel_launch(void* const* d_in, const int* in_sizes, int n_in,
                              void* d_out, int out_size, void* d_ws, size_t ws_size,
                              hipStream_t stream) {
    const float* x    = (const float*)d_in[0];
    const float* gate = (const float*)d_in[1];
    const float* bias = (const float*)d_in[2];
    const float* w1   = (const float*)d_in[3];
    const float* w2   = (const float*)d_in[4];
    const float* fc1  = (const float*)d_in[5];
    const float* fc2  = (const float*)d_in[6];
    const float* up   = (const float*)d_in[7];
    const float* down = (const float*)d_in[8];
    float* out = (float*)d_out;

    char* ws = (char*)d_ws;
    size_t o = 0;
    auto alloc = [&](size_t bytes) -> char* {
        char* p = ws + o; o += (bytes + 255) & ~size_t(255); return p;
    };
    unsigned short* xb    = (unsigned short*)alloc((size_t)NTOK * DIMD * 2);
    unsigned short* w1b   = (unsigned short*)alloc((size_t)NE * 512 * 512 * 2);
    unsigned short* w2b   = (unsigned short*)alloc((size_t)NE * 512 * 512 * 2);
    unsigned short* fc1b  = (unsigned short*)alloc((size_t)LATENTD * DIMD * 2);
    unsigned short* fc2b  = (unsigned short*)alloc((size_t)DIMD * LATENTD * 2);
    unsigned short* upb   = (unsigned short*)alloc((size_t)SHAREDD * DIMD * 2);
    unsigned short* downb = (unsigned short*)alloc((size_t)DIMD * SHAREDD * 2);
    unsigned short* latb  = (unsigned short*)alloc((size_t)NTOK * LATENTD * 2);
    int*   sel        = (int*)alloc((size_t)NSLOT * 4);
    float* wts        = (float*)alloc((size_t)NSLOT * 4);
    int*   offs       = (int*)alloc((NE + 1) * 4);
    int*   slot_tok   = (int*)alloc((size_t)NSLOT * 4);
    float* slot_score = (float*)alloc((size_t)NSLOT * 4);
    int*   tok_slot   = (int*)alloc((size_t)NSLOT * 4);
    unsigned short* xs    = (unsigned short*)alloc((size_t)NSLOT * LATENTD * 2);
    unsigned short* hbuf  = (unsigned short*)alloc((size_t)NSLOT * INTERD * 2);
    float* outslots       = (float*)alloc((size_t)NSLOT * LATENTD * 4);
    unsigned short* combb = (unsigned short*)alloc((size_t)NTOK * LATENTD * 2);
    unsigned short* shb   = (unsigned short*)alloc((size_t)NTOK * SHAREDD * 2);

    auto cvt = [&](const float* s, unsigned short* d, int n) {
        k_cvt_bf16<<<(n + 255) / 256, 256, 0, stream>>>(s, d, n);
    };
    cvt(x,    xb,    NTOK * DIMD);
    cvt(w1,   w1b,   NE * 512 * 512);
    cvt(w2,   w2b,   NE * 512 * 512);
    cvt(fc1,  fc1b,  LATENTD * DIMD);
    cvt(fc2,  fc2b,  DIMD * LATENTD);
    cvt(up,   upb,   SHAREDD * DIMD);
    cvt(down, downb, DIMD * SHAREDD);

    // router + deterministic reorder
    k_router<<<NTOK, 32, 0, stream>>>(x, gate, bias, sel, wts);
    k_sort<<<1, 32, 0, stream>>>(sel, wts, offs, slot_tok, slot_score, tok_slot);

    // lat = x @ fc1^T  -> bf16
    k_gemm<0><<<dim3(LATENTD / 128, NTOK / 64), 128, 0, stream>>>(
        xb, DIMD, fc1b, DIMD, NTOK, DIMD, LATENTD, latb, nullptr, 1.f);

    // dispatch lat rows to slots
    k_dispatch<<<(NSLOT * 256) / 256, 256, 0, stream>>>(
        (const unsigned int*)latb, slot_tok, (unsigned int*)xs);

    // grouped experts: h = relu2(xs @ w1^T), out_slots = (h @ w2^T) * score
    k_gemm_grouped<1><<<dim3(INTERD / 128, NSLOT / 64, NE), 128, 0, stream>>>(
        xs, w1b, offs, hbuf, nullptr, nullptr);
    k_gemm_grouped<4><<<dim3(LATENTD / 128, NSLOT / 64, NE), 128, 0, stream>>>(
        hbuf, w2b, offs, nullptr, outslots, slot_score);

    // combine the TOPK=2 slots per token -> bf16
    k_combine<<<(NTOK * 512) / 256, 256, 0, stream>>>(outslots, tok_slot, combb);

    // shared expert: sh = relu2(x @ up^T) ; out = sh @ down^T   (f32 store)
    k_gemm<1><<<dim3(SHAREDD / 128, NTOK / 64), 128, 0, stream>>>(
        xb, DIMD, upb, DIMD, NTOK, DIMD, SHAREDD, shb, nullptr, 1.f);
    k_gemm<2><<<dim3(DIMD / 128, NTOK / 64), 128, 0, stream>>>(
        shb, SHAREDD, downb, SHAREDD, NTOK, SHAREDD, DIMD, nullptr, out, 1.f);

    // routed = (comb @ fc2^T) * SCALE, accumulated into out
    k_gemm<3><<<dim3(DIMD / 128, NTOK / 64), 128, 0, stream>>>(
        combb, LATENTD, fc2b, LATENTD, NTOK, LATENTD, DIMD, nullptr, out, 1.0f /*SCALE*/);
}
